// LinearMultiHeadAttention_86586540687972
// MI455X (gfx1250) — matrix operand order repaired and checked by reading, not compiled, not run on hardware
//
#include <hip/hip_runtime.h>
#include <cstdint>

// ---------------- problem constants ----------------
#define B_  4
#define N_  4096
#define F_  1024
#define H_  16
#define D_  64
#define M_  (B_*N_)      // 16384 rows (b*n flattened)
#define HD_ (H_*D_)      // 1024

// ---------------- vector types ----------------
typedef __attribute__((ext_vector_type(16))) _Float16 half16;
typedef __attribute__((ext_vector_type(8)))  _Float16 half8;
typedef __attribute__((ext_vector_type(4)))  _Float16 half4;
typedef __attribute__((ext_vector_type(8)))  float    f32x8;

union Frag16 { half16 v; half8 h[2]; };

// ---------------- epilogue flavors ----------------
enum { EPI_OUT_F32 = 0,   // C = acc + bias            (final projection, f32 out)
       EPI_F16     = 1,   // C = (f16)acc              (x = q @ ctx^T)
       EPI_QSOFTMAX= 2,   // softmax over 64-col tile * D^-0.5, f16 out
       EPI_KEXP    = 3,   // C = mask? exp(acc+bias):0 (f32) + per-block column sums
       EPI_VMASK   = 4 }; // C = mask? acc+bias : 0    (f32)

struct GemmP {
  const _Float16* A;      // [M x K], row-major, stride lda
  const _Float16* BT;     // [N x K], row-major (pre-transposed B), stride ldb
  void*           C;      // output, stride ldc
  const float*    bias;   // [N] or null
  float*          kpart;  // per-block column-sum partials (EPI_KEXP)
  const unsigned char* mask; // [M] bool (EPI_KEXP / EPI_VMASK)
  int lda, ldb, ldc, K;
  int  zInnerMod;         // batch decomposition: z = outer*zInnerMod + inner
  long aOut, aInn, bOut, bInn, cOut, cInn; // element offsets per outer/inner batch idx
};

// One wave computes a 16(M) x 64(N) tile with 4x v_wmma_f32_16x16x32_f16.
// 8 waves per block stacked in M => 128x64 block tile. The 64xK B tile is
// shared by all 8 waves, so it is staged into LDS with CDNA5 async loads
// (GLOBAL_LOAD_ASYNC_TO_LDS_B128, ASYNCcnt) in double-buffered 64-K chunks.
template<int EPI>
__global__ __launch_bounds__(256) void gemm_wmma_f16(GemmP p)
{
  __shared__ _Float16 bsm[2][64][64];      // 16 KB double-buffered B chunk
  __shared__ float scol[8][64];            // only used by EPI_KEXP
  const int wave = threadIdx.x >> 5;       // wave32
  const int lane = threadIdx.x & 31;
  const int hf   = lane >> 4;              // half-wave id
  const int l15  = lane & 15;
  const int z    = blockIdx.z;
  const long aB = (long)(z / p.zInnerMod)*p.aOut + (long)(z % p.zInnerMod)*p.aInn;
  const long bB = (long)(z / p.zInnerMod)*p.bOut + (long)(z % p.zInnerMod)*p.bInn;
  const long cB = (long)(z / p.zInnerMod)*p.cOut + (long)(z % p.zInnerMod)*p.cInn;
  const int m0 = (blockIdx.x*8 + wave)*16;
  const int n0 = blockIdx.y*64;

  // A fragment: lane row = m0+l15; half 0 holds K {0..7,16..23}, half 1 {8..15,24..31}
  const _Float16* Ap = p.A + aB + (long)(m0 + l15)*p.lda + hf*8;
  const _Float16* Bbase = p.BT + bB;

  // Stage one 64(n) x 64(k) B chunk into LDS: 512 x 16B, 2 async b128 per thread.
  auto stage = [&](int c, int bf) {
#pragma unroll
    for (int u = 0; u < 2; ++u) {
      const int idx = (int)threadIdx.x + u*256;
      const int row = idx >> 3;            // 0..63 (local n)
      const int seg = idx & 7;             // 8 x 16B per 128B row
      const _Float16* g = Bbase + (long)(n0 + row)*p.ldb + c*64 + seg*8;
      const unsigned lds = (unsigned)(uintptr_t)&bsm[bf][row][seg*8];
      asm volatile("global_load_async_to_lds_b128 %0, %1, off"
                   :: "v"(lds), "v"((unsigned long long)(uintptr_t)g)
                   : "memory");
    }
  };

  f32x8 acc[4] = {};
  const int steps = p.K >> 6;              // 64-K chunks
  stage(0, 0);
  for (int c = 0; c < steps; ++c) {
    asm volatile("s_wait_asynccnt 0" ::: "memory"); // own chunk-c asyncs done
    __syncthreads();                                // => everyone's are done
    if (c + 1 < steps) stage(c + 1, (c + 1) & 1);   // overlap next stage with math
    const int bf = c & 1;
#pragma unroll
    for (int kk2 = 0; kk2 < 64; kk2 += 32) {
      const int kk = c*64 + kk2;
      Frag16 a;
      a.h[0] = *(const half8*)(Ap + kk);
      a.h[1] = *(const half8*)(Ap + kk + 16);
      __builtin_prefetch(Ap + kk + 512, 0, 0);      // -> global_prefetch_b8
#pragma unroll
      for (int t = 0; t < 4; ++t) {
        Frag16 bfrag;                               // B: half 0 holds K {0..7,16..23}, half 1 {8..15,24..31}
        bfrag.h[0] = *(const half8*)&bsm[bf][t*16 + l15][kk2 + hf*8];
        bfrag.h[1] = *(const half8*)&bsm[bf][t*16 + l15][kk2 + hf*8 + 16];
        acc[t] = __builtin_amdgcn_wmma_f32_16x16x32_f16(
            /*neg_a=*/false, a.v, /*neg_b=*/false, bfrag.v,
            /*c_mod=*/(short)0, acc[t], /*reuse_a=*/false, /*reuse_b=*/false);
      }
    }
    __syncthreads();   // all waves finished reading bf before it is restaged
  }

  // C layout: acc[t][i] -> row = m0 + i + hf*8, col = n0 + t*16 + l15
  if (EPI == EPI_QSOFTMAX) {
    // The 64-col tile is exactly one head (N tile == D == 64): softmax over cols per row.
    _Float16* C = (_Float16*)p.C + cB;
#pragma unroll
    for (int i = 0; i < 8; ++i) {
      const int row = m0 + i + hf*8;
      float vv[4]; float vmax = -1e30f;
#pragma unroll
      for (int t = 0; t < 4; ++t) {
        vv[t] = acc[t][i] + p.bias[n0 + t*16 + l15];
        vmax = fmaxf(vmax, vv[t]);
      }
#pragma unroll
      for (int m = 1; m < 16; m <<= 1) vmax = fmaxf(vmax, __shfl_xor(vmax, m, 32));
      float ee[4]; float s = 0.f;
#pragma unroll
      for (int t = 0; t < 4; ++t) { ee[t] = __expf(vv[t] - vmax); s += ee[t]; }
#pragma unroll
      for (int m = 1; m < 16; m <<= 1) s += __shfl_xor(s, m, 32);
      const float r = 0.125f / s;                // * D^-0.5
#pragma unroll
      for (int t = 0; t < 4; ++t)
        C[(long)row*p.ldc + (n0 + t*16 + l15)] = (_Float16)(ee[t]*r);
    }
  } else if (EPI == EPI_KEXP) {
    float* C = (float*)p.C + cB;
#pragma unroll
    for (int t = 0; t < 4; ++t) {
      const int col = n0 + t*16 + l15;
      const float bcol = p.bias[col];
      float cs = 0.f;
#pragma unroll
      for (int i = 0; i < 8; ++i) {
        const int row = m0 + i + hf*8;
        const float val = acc[t][i] + bcol;
        const float e = p.mask[row] ? __expf(val) : 0.f;  // masked => exp(big_neg)=0
        C[(long)row*p.ldc + col] = e;
        cs += e;
      }
      cs += __shfl_xor(cs, 16, 32);              // combine both 16-row halves
      if (hf == 0) scol[wave][t*16 + l15] = cs;
    }
    __syncthreads();
    if ((int)threadIdx.x < 64) {                 // deterministic fixed-order reduce
      float s = 0.f;
#pragma unroll
      for (int w = 0; w < 8; ++w) s += scol[w][threadIdx.x];
      p.kpart[(long)blockIdx.x*HD_ + n0 + threadIdx.x] = s;
    }
  } else if (EPI == EPI_VMASK) {
    float* C = (float*)p.C + cB;
#pragma unroll
    for (int t = 0; t < 4; ++t) {
      const int col = n0 + t*16 + l15;
      const float bcol = p.bias[col];
#pragma unroll
      for (int i = 0; i < 8; ++i) {
        const int row = m0 + i + hf*8;
        C[(long)row*p.ldc + col] = p.mask[row] ? (acc[t][i] + bcol) : 0.f;
      }
    }
  } else if (EPI == EPI_F16) {
    _Float16* C = (_Float16*)p.C + cB;
#pragma unroll
    for (int t = 0; t < 4; ++t) {
      const int col = n0 + t*16 + l15;
#pragma unroll
      for (int i = 0; i < 8; ++i) {
        const int row = m0 + i + hf*8;
        C[(long)row*p.ldc + col] = (_Float16)acc[t][i];
      }
    }
  } else { // EPI_OUT_F32
    float* C = (float*)p.C + cB;
#pragma unroll
    for (int t = 0; t < 4; ++t) {
      const int col = n0 + t*16 + l15;
      const float bcol = p.bias[col];
#pragma unroll
      for (int i = 0; i < 8; ++i) {
        const int row = m0 + i + hf*8;
        C[(long)row*p.ldc + col] = acc[t][i] + bcol;
      }
    }
  }
}

// ---------------- elementwise cast f32 -> f16 (vec4) ----------------
__global__ __launch_bounds__(256) void cast_f32_f16(const float* __restrict__ in,
                                                    _Float16* __restrict__ out, long n4)
{
  const long stride = (long)gridDim.x * blockDim.x;
  for (long i = (long)blockIdx.x*blockDim.x + threadIdx.x; i < n4; i += stride) {
    const float4 f = ((const float4*)in)[i];
    half4 o; o.x = (_Float16)f.x; o.y = (_Float16)f.y;
             o.z = (_Float16)f.z; o.w = (_Float16)f.w;
    ((half4*)out)[i] = o;
  }
}

// ---------------- 1024x1024 transpose + cast (weights) ----------------
__global__ __launch_bounds__(256) void transpose_cast_1024(const float* __restrict__ W,
                                                           _Float16* __restrict__ WT)
{
  __shared__ float tile[32][33];
  const int tx = threadIdx.x & 31, ty = threadIdx.x >> 5;
  const int f0 = blockIdx.x << 5, n0 = blockIdx.y << 5;
#pragma unroll
  for (int r = 0; r < 32; r += 8)
    tile[ty + r][tx] = W[(long)(f0 + ty + r)*1024 + (n0 + tx)];
  __syncthreads();
#pragma unroll
  for (int r = 0; r < 32; r += 8)
    WT[(long)(n0 + ty + r)*1024 + (f0 + tx)] = (_Float16)tile[tx][ty + r];
}

// ---------------- ksum: reduce 32 M-block partials per batch b ----------------
__global__ __launch_bounds__(256) void ksum_reduce(const float* __restrict__ kpart,
                                                   float* __restrict__ ksum)
{
  const int idx = blockIdx.x*256 + threadIdx.x;   // 0..4095 == b*1024 + h*64 + d
  const int b = idx >> 10, col = idx & 1023;
  float s = 0.f;
#pragma unroll 8
  for (int m = 0; m < 32; ++m) s += kpart[(long)(b*32 + m)*HD_ + col];
  ksum[idx] = s;
}

// -------- context partials: ctx[bh][d][e] += sum_n kexp[n,d]*v[n,e] over one n-chunk --------
__global__ __launch_bounds__(256) void context_partial(const float* __restrict__ kexp,
                                                       const float* __restrict__ vmsk,
                                                       float* __restrict__ part)
{
  const int bh = blockIdx.x;                 // b*16 + h
  const int chunk = blockIdx.y;              // 16 chunks x 256 n
  const int b = bh >> 4, h = bh & 15;
  __shared__ float ks[4][64], vs[4][64];
  const int t = threadIdx.x;
  const int down = t >> 2;                   // owned d (0..63)
  const int ebase = (t & 3) << 4;            // owned e range [ebase, ebase+16)
  const int r = t >> 6, dcol = t & 63;       // loader role
  float accv[16];
#pragma unroll
  for (int i = 0; i < 16; ++i) accv[i] = 0.f;
  const long base = (long)b*((long)N_*HD_) + h*64;
  for (int n0 = chunk*256; n0 < chunk*256 + 256; n0 += 4) {
    const long off = base + (long)(n0 + r)*HD_ + dcol;
    ks[r][dcol] = kexp[off];
    vs[r][dcol] = vmsk[off];
    __syncthreads();
#pragma unroll
    for (int rr = 0; rr < 4; ++rr) {
      const float kd = ks[rr][down];
#pragma unroll
      for (int i = 0; i < 16; ++i) accv[i] = fmaf(kd, vs[rr][ebase + i], accv[i]);
    }
    __syncthreads();
  }
  float* o = part + ((long)(bh*16 + chunk))*4096 + down*64 + ebase;
#pragma unroll
  for (int i = 0; i < 16; ++i) o[i] = accv[i];
}

// -------- reduce chunks, normalize by ksum, transpose -> ctxT[bh][e][d] (f16) --------
__global__ __launch_bounds__(256) void ctx_reduce(const float* __restrict__ part,
                                                  const float* __restrict__ ksum,
                                                  _Float16* __restrict__ ctxT)
{
  const int bh = blockIdx.x;
  const float* ks = ksum + (bh >> 4)*1024 + (bh & 15)*64;
  for (int idx = threadIdx.x; idx < 4096; idx += 256) {
    const int d = idx >> 6, e = idx & 63;
    float s = 0.f;
#pragma unroll
    for (int c = 0; c < 16; ++c) s += part[((long)(bh*16 + c))*4096 + idx];
    float den = ks[d]; den = (den != 0.f) ? den : 1.f;
    ctxT[(long)bh*4096 + e*64 + d] = (_Float16)(s/den);
  }
}

// ---------------- host orchestration ----------------
extern "C" void kernel_launch(void* const* d_in, const int* in_sizes, int n_in,
                              void* d_out, int out_size, void* d_ws, size_t ws_size,
                              hipStream_t stream)
{
  (void)in_sizes; (void)n_in; (void)out_size; (void)ws_size;
  const float* inq  = (const float*)d_in[0];
  const float* inkv = (const float*)d_in[1];
  const unsigned char* mask = (const unsigned char*)d_in[2]; // jnp bool: 1 byte
  const float* Wq   = (const float*)d_in[3];
  const float* bq   = (const float*)d_in[4];
  const float* Wk   = (const float*)d_in[5];
  const float* bk   = (const float*)d_in[6];
  const float* Wv   = (const float*)d_in[7];
  const float* bv   = (const float*)d_in[8];
  const float* Wout = (const float*)d_in[9];
  const float* bout = (const float*)d_in[10];
  char* ws = (char*)d_ws;

  // workspace layout (bytes)
  _Float16* inq_h  = (_Float16*)(ws + 0L);          // 32 MB
  _Float16* inkv_h = (_Float16*)(ws + 33554432L);   // 32 MB
  _Float16* WqT    = (_Float16*)(ws + 67108864L);   // 2 MB each
  _Float16* WkT    = (_Float16*)(ws + 69206016L);
  _Float16* WvT    = (_Float16*)(ws + 71303168L);
  _Float16* WoT    = (_Float16*)(ws + 73400320L);
  _Float16* q_h    = (_Float16*)(ws + 75497472L);   // 32 MB (softmaxed q, f16)
  float*    k_f    = (float*)   (ws + 109051904L);  // 64 MB (exp'd+masked k)
  float*    v_f    = (float*)   (ws + 176160768L);  // 64 MB (masked v)
  _Float16* x_h    = (_Float16*)(ws + 243269632L);  // 32 MB (attention out, f16)
  float*    kpart  = (float*)   (ws + 276824064L);  // 512 KB
  float*    ksum   = (float*)   (ws + 277348352L);  // 16 KB
  float*    cpart  = (float*)   (ws + 277364736L);  // 16 MB
  _Float16* ctxT   = (_Float16*)(ws + 294141952L);  // 512 KB

  // 1) casts / weight transposes
  cast_f32_f16<<<4096, 256, 0, stream>>>(inq,  inq_h,  (long)M_*F_/4);
  cast_f32_f16<<<4096, 256, 0, stream>>>(inkv, inkv_h, (long)M_*F_/4);
  transpose_cast_1024<<<dim3(32,32), 256, 0, stream>>>(Wq,   WqT);
  transpose_cast_1024<<<dim3(32,32), 256, 0, stream>>>(Wk,   WkT);
  transpose_cast_1024<<<dim3(32,32), 256, 0, stream>>>(Wv,   WvT);
  transpose_cast_1024<<<dim3(32,32), 256, 0, stream>>>(Wout, WoT);

  const dim3 bigGrid(M_/128, HD_/64, 1);

  // 2) Q projection + fused head-dim softmax -> q_h (f16)
  GemmP gq = { inq_h, WqT, (void*)q_h, bq, nullptr, nullptr,
               F_, F_, HD_, F_, 1, 0,0,0,0,0,0 };
  gemm_wmma_f16<EPI_QSOFTMAX><<<bigGrid, 256, 0, stream>>>(gq);

  // 3) K projection + fused mask/exp + per-block column sums
  GemmP gk = { inkv_h, WkT, (void*)k_f, bk, kpart, mask,
               F_, F_, HD_, F_, 1, 0,0,0,0,0,0 };
  gemm_wmma_f16<EPI_KEXP><<<bigGrid, 256, 0, stream>>>(gk);

  // 4) V projection + fused mask
  GemmP gv = { inkv_h, WvT, (void*)v_f, bv, nullptr, mask,
               F_, F_, HD_, F_, 1, 0,0,0,0,0,0 };
  gemm_wmma_f16<EPI_VMASK><<<bigGrid, 256, 0, stream>>>(gv);

  // 5) column softmax denominators; context = sum_n kexp*v (chunked, deterministic)
  ksum_reduce<<<16, 256, 0, stream>>>(kpart, ksum);
  context_partial<<<dim3(B_*H_, 16), 256, 0, stream>>>(k_f, v_f, cpart);
  ctx_reduce<<<B_*H_, 256, 0, stream>>>(cpart, ksum, ctxT);

  // 6) x[b,n,h,:] = q[b,n,h,:] @ ctxT[b,h]^T  (batched over 64 (b,h) pairs)
  GemmP gx = { q_h, ctxT, (void*)x_h, nullptr, nullptr, nullptr,
               HD_, D_, HD_, D_, H_,
               (long)N_*HD_, (long)D_,              // A: per-b, per-h offsets
               (long)H_*D_*D_, (long)D_*D_,         // B: ctxT batch = z*4096
               (long)N_*HD_, (long)D_ };            // C: same layout as A
  gemm_wmma_f16<EPI_F16><<<dim3(N_/128, 1, B_*H_), 256, 0, stream>>>(gx);

  // 7) output projection + bias -> d_out (f32)
  GemmP go = { x_h, WoT, d_out, bout, nullptr, nullptr,
               HD_, HD_, F_, HD_, 1, 0,0,0,0,0,0 };
  gemm_wmma_f16<EPI_OUT_F32><<<dim3(M_/128, F_/64, 1), 256, 0, stream>>>(go);
}
